// MLLAFormer_8804682957521
// MI455X (gfx1250) — matrix operand order repaired and checked by reading, not compiled, not run on hardware
//
#include <hip/hip_runtime.h>
#include <hip/hip_bf16.h>
#include <math.h>

// ---------------------------------------------------------------------------
// MLLAFormer block for MI455X (gfx1250, wave32, WMMA 16x16x32 f16).
// GEMMs stage f32->f16 tiles in *fragment order* LDS so a fragment is a
// single v16h (2x ds_load_b128); staging uses float4 global loads, packed
// cvt and b64 LDS stores; K-loop is double-buffered. Elementwise kernels
// are float4-vectorized.
// ---------------------------------------------------------------------------

typedef __attribute__((ext_vector_type(16))) _Float16 v16h;
typedef __attribute__((ext_vector_type(4)))  _Float16 v4h;
typedef __attribute__((ext_vector_type(8)))  float    v8f;

#define C_DIM   192
#define HID_DIM 768
#define N_HEADS 6
#define D_HEAD  32
#define NSP     4096            // 64*64 spatial positions per batch image
#define NBATCH  16
#define MROWS   (NBATCH * NSP)  // 65536 total tokens

// A-operand (16x32 f16) lane striping:  lane = hi*16 + (m&15),
//   j(0..7) <-> k = hi*8 + j ; j(8..15) <-> k = 8 + hi*8 + j
__device__ __forceinline__ int a_hi(int k) { return (k >> 3) & 1; }
__device__ __forceinline__ int a_j (int k) { return (k & 7) | ((k & 16) >> 1); }
// B-operand (32x16 f16) lane striping:  same as A, lane = hi*16 + (n&15), j = a_j(k)

// -------------------------------- activations ------------------------------
template <int ACT>
__device__ __forceinline__ float act_fn(float v) {
    if (ACT == 1) return v / (1.f + __expf(-v));                       // silu
    if (ACT == 2) return 0.5f * v * (1.f + erff(v * 0.70710678118f));  // gelu (exact)
    if (ACT == 3) return v > 0.f ? v + 1.f : __expf(v);                // elu+1
    return v;
}

// ------------------------ generic token-major WMMA GEMM --------------------
// out = ACT( A[M,K] @ W[N,K]^T + bias )    (A token-major f32, W row-major f32)
// EPI==0 : token-major out[M,N];  EPI==1 : NCHW + residual
template <int ACT, int EPI>
__global__ __launch_bounds__(256) void gemm_wmma_kernel(
    const float* __restrict__ A, const float* __restrict__ W,
    const float* __restrict__ bias, const float* __restrict__ resid,
    float* __restrict__ out, int M, int K, int N)
{
    constexpr int BM = 128, BN = 64, BK = 32;
    constexpr int ASZ = (BM / 16) * 512;   // 4096 halves, fragment-order
    constexpr int BSZ = (BN / 16) * 512;   // 2048 halves
    __shared__ __align__(32) _Float16 sA[2 * ASZ];
    __shared__ __align__(32) _Float16 sB[2 * BSZ];

    const int tid  = threadIdx.x;
    const int lane = tid & 31, wid = tid >> 5;
    const int n0 = blockIdx.x * BN, m0 = blockIdx.y * BM;
    const int hi = lane >> 4, nl = lane & 15;

    float4 ra[4], rb[2];                    // register prefetch buffers

    auto gload = [&](int k0) {
#pragma unroll
        for (int i = 0; i < 4; ++i) {       // A: 1024 quads / 256 thr
            int q = tid + 256 * i;
            int m = q >> 3, kq = (q & 7) * 4;
            ra[i] = *(const float4*)&A[(size_t)(m0 + m) * K + k0 + kq];
        }
#pragma unroll
        for (int i = 0; i < 2; ++i) {       // B: 512 quads / 256 thr
            int q = tid + 256 * i;
            int n = q >> 3, kq = (q & 7) * 4;
            rb[i] = *(const float4*)&W[(size_t)(n0 + n) * K + k0 + kq];
        }
    };
    auto stage = [&](int buf) {
#pragma unroll
        for (int i = 0; i < 4; ++i) {
            int q = tid + 256 * i;
            int m = q >> 3, kq = (q & 7) * 4;
            v4h hv; hv[0] = (_Float16)ra[i].x; hv[1] = (_Float16)ra[i].y;
            hv[2] = (_Float16)ra[i].z; hv[3] = (_Float16)ra[i].w;
            *(v4h*)&sA[buf * ASZ + (m >> 4) * 512 +
                       (a_hi(kq) * 16 + (m & 15)) * 16 + a_j(kq)] = hv;
        }
#pragma unroll
        for (int i = 0; i < 2; ++i) {
            int q = tid + 256 * i;
            int n = q >> 3, kq = (q & 7) * 4;
            v4h hv; hv[0] = (_Float16)rb[i].x; hv[1] = (_Float16)rb[i].y;
            hv[2] = (_Float16)rb[i].z; hv[3] = (_Float16)rb[i].w;
            *(v4h*)&sB[buf * BSZ + (n >> 4) * 512 +
                       (a_hi(kq) * 16 + (n & 15)) * 16 + a_j(kq)] = hv;
        }
    };

    v8f acc[4] = {};
    const int nk = K / BK;
    gload(0);
    int buf = 0;
    for (int it = 0; it < nk; ++it) {
        stage(buf);
        __syncthreads();
        if (it + 1 < nk) gload((it + 1) * BK);    // overlap next tile
        v16h afrag = *(const v16h*)&sA[buf * ASZ + wid * 512 + lane * 16];
#pragma unroll
        for (int nt = 0; nt < 4; ++nt) {
            v16h bfrag = *(const v16h*)&sB[buf * BSZ + nt * 512 + lane * 16];
            acc[nt] = __builtin_amdgcn_wmma_f32_16x16x32_f16(
                false, afrag, false, bfrag, (short)0, acc[nt], false, false);
        }
        buf ^= 1;
    }

#pragma unroll
    for (int nt = 0; nt < 4; ++nt) {
        int col = n0 + nt * 16 + nl;
        float bv = bias[col];
#pragma unroll
        for (int r = 0; r < 8; ++r) {
            int row = m0 + wid * 16 + hi * 8 + r;
            float val = act_fn<ACT>(acc[nt][r] + bv);
            if (EPI == 0) {
                out[(size_t)row * N + col] = val;
            } else {
                int bimg = row >> 12, hw = row & 4095;
                size_t idx = ((size_t)bimg * N + col) * NSP + hw;
                out[idx] = resid[idx] + val;
            }
        }
    }
}

// --------------------------- depthwise conv kernels ------------------------
// x = x + dwconv3x3(x), NCHW layout (hw contiguous -> coalesced scalar)
__global__ __launch_bounds__(256) void dwconv_res_nchw_kernel(
    const float* __restrict__ x, const float* __restrict__ w,
    const float* __restrict__ bias, float* __restrict__ out)
{
    size_t idx = (size_t)blockIdx.x * 256 + threadIdx.x;     // B*C*4096
    int hw = (int)(idx & 4095);
    size_t bc = idx >> 12;
    int c = (int)(bc % C_DIM);
    int h = hw >> 6, wc = hw & 63;
    const float* xp = x + (bc << 12);
    float acc = bias[c];
#pragma unroll
    for (int dy = -1; dy <= 1; ++dy) {
        int nh = h + dy; if ((unsigned)nh >= 64u) continue;
#pragma unroll
        for (int dx = -1; dx <= 1; ++dx) {
            int nw = wc + dx; if ((unsigned)nw >= 64u) continue;
            acc += xp[nh * 64 + nw] * w[c * 9 + (dy + 1) * 3 + (dx + 1)];
        }
    }
    out[idx] = x[idx] + acc;
}

// t = silu(dwconv3x3(y)+b), token-major, 4 channels per thread (float4)
__global__ __launch_bounds__(256) void dwc_silu_kernel(
    const float* __restrict__ y, const float* __restrict__ w,
    const float* __restrict__ bias, float* __restrict__ t)
{
    size_t idx4 = (size_t)blockIdx.x * 256 + threadIdx.x;    // MROWS*48
    int cq = (int)(idx4 % 48) * 4;
    size_t tok = idx4 / 48;
    int hw = (int)(tok & 4095), bimg = (int)(tok >> 12);
    int h = hw >> 6, wc = hw & 63;
    float4 acc = *(const float4*)&bias[cq];
#pragma unroll
    for (int dy = -1; dy <= 1; ++dy) {
        int nh = h + dy; if ((unsigned)nh >= 64u) continue;
#pragma unroll
        for (int dx = -1; dx <= 1; ++dx) {
            int nw = wc + dx; if ((unsigned)nw >= 64u) continue;
            float4 s = *(const float4*)&y[((size_t)(bimg << 12) + nh * 64 + nw) * C_DIM + cq];
            int kk = (dy + 1) * 3 + (dx + 1);
            acc.x += s.x * w[(cq + 0) * 9 + kk];
            acc.y += s.y * w[(cq + 1) * 9 + kk];
            acc.z += s.z * w[(cq + 2) * 9 + kk];
            acc.w += s.w * w[(cq + 3) * 9 + kk];
        }
    }
    float4 o;
    o.x = acc.x / (1.f + __expf(-acc.x));
    o.y = acc.y / (1.f + __expf(-acc.y));
    o.z = acc.z / (1.f + __expf(-acc.z));
    o.w = acc.w / (1.f + __expf(-acc.w));
    *(float4*)&t[tok * C_DIM + cq] = o;
}

// h = (attn + lepe(t)) * act_res, token-major, 4 channels per thread
__global__ __launch_bounds__(256) void lepe_had_kernel(
    const float* __restrict__ attn, const float* __restrict__ t,
    const float* __restrict__ w, const float* __restrict__ bias,
    const float* __restrict__ actres, float* __restrict__ outh)
{
    size_t idx4 = (size_t)blockIdx.x * 256 + threadIdx.x;    // MROWS*48
    int cq = (int)(idx4 % 48) * 4;
    size_t tok = idx4 / 48;
    int hw = (int)(tok & 4095), bimg = (int)(tok >> 12);
    int h = hw >> 6, wc = hw & 63;
    float4 acc = *(const float4*)&bias[cq];
#pragma unroll
    for (int dy = -1; dy <= 1; ++dy) {
        int nh = h + dy; if ((unsigned)nh >= 64u) continue;
#pragma unroll
        for (int dx = -1; dx <= 1; ++dx) {
            int nw = wc + dx; if ((unsigned)nw >= 64u) continue;
            float4 s = *(const float4*)&t[((size_t)(bimg << 12) + nh * 64 + nw) * C_DIM + cq];
            int kk = (dy + 1) * 3 + (dx + 1);
            acc.x += s.x * w[(cq + 0) * 9 + kk];
            acc.y += s.y * w[(cq + 1) * 9 + kk];
            acc.z += s.z * w[(cq + 2) * 9 + kk];
            acc.w += s.w * w[(cq + 3) * 9 + kk];
        }
    }
    float4 a = *(const float4*)&attn[tok * C_DIM + cq];
    float4 g = *(const float4*)&actres[tok * C_DIM + cq];
    float4 o;
    o.x = (a.x + acc.x) * g.x;
    o.y = (a.y + acc.y) * g.y;
    o.z = (a.z + acc.z) * g.z;
    o.w = (a.w + acc.w) * g.w;
    *(float4*)&outh[tok * C_DIM + cq] = o;
}

// --------------------- LayerNorm over C + NCHW -> token-major --------------
__global__ __launch_bounds__(192) void ln_transpose_kernel(
    const float* __restrict__ x, const float* __restrict__ g,
    const float* __restrict__ b2, float* __restrict__ out)
{
    const int sp = blockIdx.x;                 // token id
    const int bimg = sp >> 12, hw = sp & 4095, c = threadIdx.x;
    float v = x[((size_t)bimg * C_DIM + c) * NSP + hw];
    __shared__ float wsum[6], wsum2[6];
    __shared__ float mu_s, rsig_s;
    float s = v, s2 = v * v;
#pragma unroll
    for (int off = 16; off > 0; off >>= 1) {
        s  += __shfl_down(s, off);
        s2 += __shfl_down(s2, off);
    }
    int lane = threadIdx.x & 31, w = threadIdx.x >> 5;
    if (lane == 0) { wsum[w] = s; wsum2[w] = s2; }
    __syncthreads();
    if (threadIdx.x == 0) {
        float t = 0.f, t2 = 0.f;
        for (int i = 0; i < 6; ++i) { t += wsum[i]; t2 += wsum2[i]; }
        float mu = t * (1.f / C_DIM);
        float var = t2 * (1.f / C_DIM) - mu * mu;
        mu_s = mu; rsig_s = rsqrtf(var + 1e-5f);
    }
    __syncthreads();
    out[(size_t)sp * C_DIM + c] = (v - mu_s) * rsig_s * g[c] + b2[c];
}

// ----------------------------- attention pieces ----------------------------
__global__ __launch_bounds__(256) void kmean_kernel(
    const float* __restrict__ qk, float* __restrict__ kmean)
{
    int bc = blockIdx.x;                      // b*192 + c
    int bimg = bc / C_DIM, c = bc % C_DIM;
    float s = 0.f;
    for (int n = threadIdx.x; n < NSP; n += 256)
        s += qk[((size_t)(bimg << 12) + n) * 384 + 192 + c];
    __shared__ float red[256];
    red[threadIdx.x] = s; __syncthreads();
    for (int off = 128; off > 0; off >>= 1) {
        if (threadIdx.x < off) red[threadIdx.x] += red[threadIdx.x + off];
        __syncthreads();
    }
    if (threadIdx.x == 0) kmean[bc] = red[0] * (1.f / NSP);
}

__global__ __launch_bounds__(256) void z_kernel(
    const float* __restrict__ qk, const float* __restrict__ kmean,
    float* __restrict__ z)
{
    size_t idx = (size_t)blockIdx.x * 256 + threadIdx.x;     // MROWS*6
    int hh = (int)(idx % N_HEADS);
    size_t tok = idx / N_HEADS;
    int bimg = (int)(tok >> 12);
    float dot = 0.f;
#pragma unroll
    for (int kd = 0; kd < D_HEAD; ++kd)
        dot += qk[tok * 384 + hh * 32 + kd] * kmean[bimg * C_DIM + hh * 32 + kd];
    z[idx] = 1.f / (dot + 1e-6f);
}

// in-place RoPE; each thread rotates 2 (re,im) pairs of q and k via float4
__global__ __launch_bounds__(256) void rope_kernel(float* __restrict__ qk)
{
    size_t idx = (size_t)blockIdx.x * 256 + threadIdx.x;     // MROWS*48
    int jg = (int)(idx % 48);
    size_t tok = idx / 48;
    int hw = (int)(tok & 4095);
    int h = hw >> 6, wc = hw & 63;
    int p0 = 2 * jg, p1 = p0 + 1;
    int i0 = (p0 < 48) ? p0 : p0 - 48;
    int i1 = (p1 < 48) ? p1 : p1 - 48;
    float pos = (p0 < 48) ? (float)h : (float)wc;
    float th0 = __expf(-9.2103403720f * ((float)i0 * (1.f / 48.f)));
    float th1 = __expf(-9.2103403720f * ((float)i1 * (1.f / 48.f)));
    float s0, c0, s1, c1;
    __sincosf(pos * th0, &s0, &c0);
    __sincosf(pos * th1, &s1, &c1);

    float4* qp = (float4*)(qk + tok * 384 + 4 * jg);
    float4 v = *qp;
    *qp = make_float4(v.x * c0 - v.y * s0, v.x * s0 + v.y * c0,
                      v.z * c1 - v.w * s1, v.z * s1 + v.w * c1);
    float4* kp = (float4*)(qk + tok * 384 + 192 + 4 * jg);
    v = *kp;
    *kp = make_float4(v.x * c0 - v.y * s0, v.x * s0 + v.y * c0,
                      v.z * c1 - v.w * s1, v.z * s1 + v.w * c1);
}

// kv[b,h] (32x32) = sum_n (k_rope/64)^T (v/64), K=4096 WMMA reduction.
// Staging: one thread handles a token-quad for one channel -> v4h LDS stores
// (ds_store_b64) with coalesced global reads; deterministic tree reduction.
__global__ __launch_bounds__(256) void kv_kernel(
    const float* __restrict__ qk, const float* __restrict__ t,
    float* __restrict__ kv)
{
    const int bh = blockIdx.x;                // b*6 + h
    const int bimg = bh / N_HEADS, hh = bh % N_HEADS;
    const int tid = threadIdx.x;
    const int lane = tid & 31, wid = tid >> 5;
    const int hi = lane >> 4, nl = lane & 15;
    const float scale = 0.015625f;            // 1/sqrt(4096)
    const size_t tokbase = (size_t)bimg << 12;

    __shared__ __align__(32) char smem[32768];
    _Float16* sKA = (_Float16*)smem;            // 8 chunks x 1024 halves
    _Float16* sKV = (_Float16*)(smem + 16384);  // 8 chunks x 1024 halves
    float*    sW  = (float*)smem;               // aliased after compute

    v8f acc[2][2] = {};
    for (int grp = 0; grp < 16; ++grp) {       // 16 groups x 8 chunks x 32 tok
        __syncthreads();
#pragma unroll
        for (int i = 0; i < 8; ++i) {
            int q = tid + 256 * i;             // 0..2047 (m, token-quad)
            int m = q & 31;                    // channel within head
            int kqg = (q >> 5) * 4;            // token quad within group
            int chl = kqg >> 5, kl = kqg & 31;
            size_t tok = tokbase + (size_t)grp * 256 + kqg;
            v4h ha, hb;
#pragma unroll
            for (int u = 0; u < 4; ++u) {
                ha[u] = (_Float16)(scale * qk[(tok + u) * 384 + 192 + hh * 32 + m]);
                hb[u] = (_Float16)(scale * t[(tok + u) * C_DIM + hh * 32 + m]);
            }
            *(v4h*)&sKA[chl * 1024 + (m >> 4) * 512 +
                        (a_hi(kl) * 16 + (m & 15)) * 16 + a_j(kl)] = ha;
            *(v4h*)&sKV[chl * 1024 + (m >> 4) * 512 +
                        (a_hi(kl) * 16 + (m & 15)) * 16 + a_j(kl)] = hb;
        }
        __syncthreads();
        v16h af0 = *(const v16h*)&sKA[wid * 1024 + lane * 16];
        v16h af1 = *(const v16h*)&sKA[wid * 1024 + 512 + lane * 16];
        v16h bf0 = *(const v16h*)&sKV[wid * 1024 + lane * 16];
        v16h bf1 = *(const v16h*)&sKV[wid * 1024 + 512 + lane * 16];
        acc[0][0] = __builtin_amdgcn_wmma_f32_16x16x32_f16(false, af0, false, bf0, (short)0, acc[0][0], false, false);
        acc[0][1] = __builtin_amdgcn_wmma_f32_16x16x32_f16(false, af0, false, bf1, (short)0, acc[0][1], false, false);
        acc[1][0] = __builtin_amdgcn_wmma_f32_16x16x32_f16(false, af1, false, bf0, (short)0, acc[1][0], false, false);
        acc[1][1] = __builtin_amdgcn_wmma_f32_16x16x32_f16(false, af1, false, bf1, (short)0, acc[1][1], false, false);
    }
    __syncthreads();
    // per-wave partials -> deterministic tree sum
#pragma unroll
    for (int mt = 0; mt < 2; ++mt)
#pragma unroll
        for (int nt = 0; nt < 2; ++nt)
#pragma unroll
            for (int r = 0; r < 8; ++r)
                sW[wid * 1024 + (mt * 16 + hi * 8 + r) * 32 + nt * 16 + nl] =
                    acc[mt][nt][r];
    __syncthreads();
    for (int i = tid; i < 1024; i += 256) {
        float s = 0.f;
#pragma unroll
        for (int w = 0; w < 8; ++w) s += sW[w * 1024 + i];
        kv[(size_t)bh * 1024 + i] = s;
    }
}

// attn[tok, h*32+e] = (q_rope[tok,h,:] @ kv[b,h]) * z[tok,h]
__global__ __launch_bounds__(256) void attn_kernel(
    const float* __restrict__ qk, const float* __restrict__ kv,
    const float* __restrict__ z, float* __restrict__ attn)
{
    const int blk = blockIdx.x;               // (b*6+h)*32 + token-chunk
    const int bh = blk >> 5, tch = blk & 31;
    const int bimg = bh / N_HEADS, hh = bh % N_HEADS;
    const int tid = threadIdx.x;
    const int lane = tid & 31, wid = tid >> 5;
    const int mrow = lane & 15, hi = lane >> 4, nl = lane & 15;

    __shared__ __align__(32) _Float16 skv[2 * 512];   // fragment-order kv
    for (int i = tid; i < 1024; i += 256) {
        int k = i >> 5, n = i & 31;
        skv[(n >> 4) * 512 + (a_hi(k) * 16 + (n & 15)) * 16 + a_j(k)] =
            (_Float16)kv[(size_t)bh * 1024 + i];
    }
    __syncthreads();

    const size_t t0 = ((size_t)bimg << 12) + (size_t)tch * 128 + wid * 16;
    // q_rope fragment = two contiguous 8-float spans per lane
    const float* qrow = &qk[(t0 + mrow) * 384 + hh * 32 + hi * 8];
    float4 f0 = *(const float4*)qrow;
    float4 f1 = *(const float4*)(qrow + 4);
    float4 f2 = *(const float4*)(qrow + 16);
    float4 f3 = *(const float4*)(qrow + 20);
    v16h af;
    af[0]  = (_Float16)f0.x; af[1]  = (_Float16)f0.y; af[2]  = (_Float16)f0.z; af[3]  = (_Float16)f0.w;
    af[4]  = (_Float16)f1.x; af[5]  = (_Float16)f1.y; af[6]  = (_Float16)f1.z; af[7]  = (_Float16)f1.w;
    af[8]  = (_Float16)f2.x; af[9]  = (_Float16)f2.y; af[10] = (_Float16)f2.z; af[11] = (_Float16)f2.w;
    af[12] = (_Float16)f3.x; af[13] = (_Float16)f3.y; af[14] = (_Float16)f3.z; af[15] = (_Float16)f3.w;

    v8f acc[2] = {};
#pragma unroll
    for (int nt = 0; nt < 2; ++nt) {
        v16h bf = *(const v16h*)&skv[nt * 512 + lane * 16];
        acc[nt] = __builtin_amdgcn_wmma_f32_16x16x32_f16(
            false, af, false, bf, (short)0, acc[nt], false, false);
    }
#pragma unroll
    for (int nt = 0; nt < 2; ++nt) {
#pragma unroll
        for (int r = 0; r < 8; ++r) {
            size_t tok = t0 + hi * 8 + r;
            float zr = z[tok * N_HEADS + hh];
            attn[tok * C_DIM + hh * 32 + nt * 16 + nl] = acc[nt][r] * zr;
        }
    }
}

// --------------------------------- launch ----------------------------------
extern "C" void kernel_launch(void* const* d_in, const int* in_sizes, int n_in,
                              void* d_out, int out_size, void* d_ws, size_t ws_size,
                              hipStream_t stream)
{
    (void)in_sizes; (void)n_in; (void)out_size; (void)ws_size;
    const float* x          = (const float*)d_in[0];
    const float* cpe1_w     = (const float*)d_in[1];
    const float* cpe1_b     = (const float*)d_in[2];
    const float* norm1_w    = (const float*)d_in[3];
    const float* norm1_b    = (const float*)d_in[4];
    const float* act_proj_w = (const float*)d_in[5];
    const float* act_proj_b = (const float*)d_in[6];
    const float* in_proj_w  = (const float*)d_in[7];
    const float* in_proj_b  = (const float*)d_in[8];
    const float* dwc_w      = (const float*)d_in[9];
    const float* dwc_b      = (const float*)d_in[10];
    const float* qk_w       = (const float*)d_in[11];
    const float* qk_b       = (const float*)d_in[12];
    const float* lepe_w     = (const float*)d_in[13];
    const float* lepe_b     = (const float*)d_in[14];
    const float* out_proj_w = (const float*)d_in[15];
    const float* out_proj_b = (const float*)d_in[16];
    const float* cpe2_w     = (const float*)d_in[17];
    const float* cpe2_b     = (const float*)d_in[18];
    const float* norm2_w    = (const float*)d_in[19];
    const float* norm2_b    = (const float*)d_in[20];
    const float* fc1_w      = (const float*)d_in[21];
    const float* fc1_b      = (const float*)d_in[22];
    const float* fc2_w      = (const float*)d_in[23];
    const float* fc2_b      = (const float*)d_in[24];
    float* out = (float*)d_out;

    const size_t SZ = (size_t)MROWS * C_DIM;
    float* ws    = (float*)d_ws;
    float* buf0  = ws;                 // x1 (NCHW)
    float* buf1  = buf0  + SZ;         // ln1 / hbuf / ln2
    float* buf2  = buf1  + SZ;         // act_res, then x2 (NCHW)
    float* buf3  = buf2  + SZ;         // y (in_proj), then attn
    float* buft  = buf3  + SZ;         // t = v
    float* bufqk = buft  + SZ;         // qk (2*SZ)
    float* buf4  = bufqk + 2 * SZ;     // MLP hidden (4*SZ)
    float* kmean = buf4  + 4 * SZ;     // 16*192
    float* zbuf  = kmean + NBATCH * C_DIM;          // MROWS*6
    float* kvb   = zbuf  + (size_t)MROWS * N_HEADS; // 96*1024

    const int ELEM_BLKS  = (int)(SZ / 256);         // 49152 (scalar elementwise)
    const int ELEM4_BLKS = (int)(SZ / 4 / 256);     // 12288 (float4 elementwise)

    // 1) x1 = x + cpe1(x)
    dwconv_res_nchw_kernel<<<ELEM_BLKS, 256, 0, stream>>>(x, cpe1_w, cpe1_b, buf0);
    // 2) ln1 -> token-major
    ln_transpose_kernel<<<MROWS, 192, 0, stream>>>(buf0, norm1_w, norm1_b, buf1);
    // 3) act_res = silu(ln1 @ act_proj^T + b)
    gemm_wmma_kernel<1, 0><<<dim3(C_DIM / 64, MROWS / 128), 256, 0, stream>>>(
        buf1, act_proj_w, act_proj_b, nullptr, buf2, MROWS, C_DIM, C_DIM);
    // 4) y = ln1 @ in_proj^T + b
    gemm_wmma_kernel<0, 0><<<dim3(C_DIM / 64, MROWS / 128), 256, 0, stream>>>(
        buf1, in_proj_w, in_proj_b, nullptr, buf3, MROWS, C_DIM, C_DIM);
    // 5) t = silu(dwc(y))
    dwc_silu_kernel<<<ELEM4_BLKS, 256, 0, stream>>>(buf3, dwc_w, dwc_b, buft);
    // 6) qk = elu(t @ qk_w^T + b) + 1
    gemm_wmma_kernel<3, 0><<<dim3(384 / 64, MROWS / 128), 256, 0, stream>>>(
        buft, qk_w, qk_b, nullptr, bufqk, MROWS, C_DIM, 2 * C_DIM);
    // 7) kmean over tokens
    kmean_kernel<<<NBATCH * C_DIM, 256, 0, stream>>>(bufqk, kmean);
    // 8) z = 1/(q.kmean + eps)
    z_kernel<<<MROWS * N_HEADS / 256, 256, 0, stream>>>(bufqk, kmean, zbuf);
    // 9) RoPE in-place on q,k
    rope_kernel<<<MROWS * 48 / 256, 256, 0, stream>>>(bufqk);
    // 10) kv = (k_rope*s)^T (v*s) per (b,h)
    kv_kernel<<<NBATCH * N_HEADS, 256, 0, stream>>>(bufqk, buft, kvb);
    // 11) attn = (q_rope @ kv) * z
    attn_kernel<<<NBATCH * N_HEADS * 32, 256, 0, stream>>>(bufqk, kvb, zbuf, buf3);
    // 12) hbuf = (attn + lepe(v)) * act_res
    lepe_had_kernel<<<ELEM4_BLKS, 256, 0, stream>>>(buf3, buft, lepe_w, lepe_b, buf2, buf1);
    // 13) x2 = x1 + out_proj(hbuf)
    gemm_wmma_kernel<0, 1><<<dim3(C_DIM / 64, MROWS / 128), 256, 0, stream>>>(
        buf1, out_proj_w, out_proj_b, buf0, buf2, MROWS, C_DIM, C_DIM);
    // 14) x3 = x2 + cpe2(x2)  -> d_out
    dwconv_res_nchw_kernel<<<ELEM_BLKS, 256, 0, stream>>>(buf2, cpe2_w, cpe2_b, out);
    // 15) ln2 -> token-major
    ln_transpose_kernel<<<MROWS, 192, 0, stream>>>(out, norm2_w, norm2_b, buf1);
    // 16) hid = gelu(ln2 @ fc1^T + b)
    gemm_wmma_kernel<2, 0><<<dim3(HID_DIM / 64, MROWS / 128), 256, 0, stream>>>(
        buf1, fc1_w, fc1_b, nullptr, buf4, MROWS, C_DIM, HID_DIM);
    // 17) out = x3 + fc2(hid)
    gemm_wmma_kernel<0, 1><<<dim3(C_DIM / 64, MROWS / 128), 256, 0, stream>>>(
        buf4, fc2_w, fc2_b, out, out, MROWS, HID_DIM, C_DIM);
}
